// PerceiverEncoder_50319836840091
// MI455X (gfx1250) — matrix operand order repaired and checked by reading, not compiled, not run on hardware
//
#include <hip/hip_runtime.h>
#include <hip/hip_bf16.h>
#include <math.h>

// ---------------------------------------------------------------------------
// MI455X (gfx1250) Perceiver encoder.
// Memory-bound (~0.93 TFLOP vs ~450MB traffic @23.3TB/s): fp32 operands are
// converted to f16 on the fly into LDS (packed b128/b32 stores), all matmuls
// run on v_wmma_f32_16x16x32_f16 with f32 accumulate, attention is flash-style
// so scores never touch HBM, and both hot loops are LDS double-buffered so
// global loads overlap the WMMA stream.
// ---------------------------------------------------------------------------

typedef __attribute__((ext_vector_type(16))) _Float16 v16h;
typedef __attribute__((ext_vector_type(8)))  _Float16 v8h;
typedef __attribute__((ext_vector_type(8)))  float    v8f;
typedef __attribute__((ext_vector_type(4)))  float    v4f;

union V16U { v16h v; v8h h[2]; _Float16 e[16]; };
union PK2  { _Float16 h[2]; unsigned u; };

__device__ __forceinline__ unsigned pk2(float lo, float hi) {
    PK2 t; t.h[0] = (_Float16)lo; t.h[1] = (_Float16)hi; return t.u;
}

// Build a 16x f16 fragment from two 16B-aligned 8-element halves.
__device__ __forceinline__ v16h mk16(const _Float16* p0, const _Float16* p1) {
    V16U u;
    u.h[0] = *(const v8h*)p0;
    u.h[1] = *(const v8h*)p1;
    return u.v;
}

__device__ __forceinline__ v8f wmma16(v16h a, v16h b, v8f c) {
    // (neg_a, A, neg_b, B, c_mod, C, reuse_a, reuse_b)
    return __builtin_amdgcn_wmma_f32_16x16x32_f16(false, a, false, b,
                                                  (short)0, c, false, false);
}

__device__ __forceinline__ v8f v8f_zero() {
    v8f z;
#pragma unroll
    for (int i = 0; i < 8; ++i) z[i] = 0.0f;
    return z;
}

// ---------------------------------------------------------------------------
// Broadcast latent [N*D] -> h [B, N*D]
// ---------------------------------------------------------------------------
__global__ void bcast_kernel(const float* __restrict__ lat,
                             float* __restrict__ h, int nd, int total) {
    int i = blockIdx.x * 256 + threadIdx.x;
    if (i < total) h[i] = lat[i % nd];
}

// ---------------------------------------------------------------------------
// Row LayerNorm (eps 1e-5), one block (256 thr / 8 waves) per row.
// ---------------------------------------------------------------------------
__global__ __launch_bounds__(256, 1)
void ln_kernel(const float* __restrict__ in, float* __restrict__ out,
               const float* __restrict__ w, const float* __restrict__ b,
               int dim) {
    __shared__ float red0[8];
    __shared__ float red1[8];
    const size_t row = blockIdx.x;
    const float* x = in + row * (size_t)dim;
    float s = 0.f, s2 = 0.f;
    for (int i = threadIdx.x; i < dim; i += 256) {
        float v = x[i];
        s += v; s2 += v * v;
    }
#pragma unroll
    for (int off = 1; off < 32; off <<= 1) {
        s  += __shfl_xor(s,  off, 32);
        s2 += __shfl_xor(s2, off, 32);
    }
    const int wid = threadIdx.x >> 5, lane = threadIdx.x & 31;
    if (lane == 0) { red0[wid] = s; red1[wid] = s2; }
    __syncthreads();
    if (threadIdx.x == 0) {
        float a = 0.f, c = 0.f;
        for (int i = 0; i < 8; ++i) { a += red0[i]; c += red1[i]; }
        red0[0] = a; red1[0] = c;
    }
    __syncthreads();
    const float inv = 1.0f / (float)dim;
    const float mu  = red0[0] * inv;
    const float var = red1[0] * inv - mu * mu;
    const float rs  = rsqrtf(var + 1e-5f);
    float* y = out + row * (size_t)dim;
    for (int i = threadIdx.x; i < dim; i += 256)
        y[i] = (x[i] - mu) * rs * w[i] + b[i];
}

// ---------------------------------------------------------------------------
// WMMA GEMM: out[M,N] = A[M,K] @ W[K,N] + bias (+res) (opt exact GELU)
// Block tile 64x256 (8 waves, each 32x64 = 2x4 WMMA tiles), K-step 32.
// Double-buffered LDS; packed f16 staging stores (b128 for A, b32 pairs for W).
// Requires M%64==0, N%256==0, K%32==0 (true for all shapes here).
// ---------------------------------------------------------------------------
#define G_LAS 40                // lA row stride (f16): 80B, 16B-multiple
#define G_LBS 40                // lBt row stride (f16)
#define G_ASZ (64 * G_LAS)      // one A buffer
#define G_BSZ (256 * G_LBS)     // one B buffer

__global__ __launch_bounds__(256, 1)
void gemm_kernel(const float* __restrict__ A, const float* __restrict__ W,
                 const float* __restrict__ bias, const float* __restrict__ res,
                 float* __restrict__ out, int K, int Nn, int gelu) {
    __shared__ _Float16 lA[2 * G_ASZ];
    __shared__ _Float16 lB[2 * G_BSZ];   // transposed: [n][k]

    const int tid   = threadIdx.x;
    const int lane  = tid & 31;
    const int wid   = tid >> 5;
    const int half  = lane >> 4;
    const int l16   = lane & 15;
    const int waveM = wid >> 2;            // 0..1
    const int waveN = wid & 3;             // 0..3
    const int row0  = blockIdx.y * 64;
    const int col0  = blockIdx.x * 256;

    v8f acc[2][4];
#pragma unroll
    for (int i = 0; i < 2; ++i)
#pragma unroll
        for (int j = 0; j < 4; ++j) acc[i][j] = v8f_zero();

    const int ar = tid >> 2;               // 0..63
    const int ac = (tid & 3) * 8;          // 0..24
    const int bk = tid >> 5;               // 0..7  -> k pairs bk, bk+8
    const int bn = (tid & 31) * 8;         // 0..248

    // staged raw tiles (kept as loaded vectors so loadcnt-wait lands in store)
    v4f rA0, rA1;
    v4f rB[2][4];                          // [rep][row0 lo/hi, row1 lo/hi]

    auto g_load = [&](int k0) {
        const float* ap = A + (size_t)(row0 + ar) * K + k0 + ac;
        rA0 = *(const v4f*)ap;
        rA1 = *(const v4f*)(ap + 4);
#pragma unroll
        for (int rep = 0; rep < 2; ++rep) {
            const int k = 2 * (bk + rep * 8);
            const float* wp = W + (size_t)(k0 + k) * Nn + col0 + bn;
            rB[rep][0] = *(const v4f*)wp;
            rB[rep][1] = *(const v4f*)(wp + 4);
            rB[rep][2] = *(const v4f*)(wp + Nn);
            rB[rep][3] = *(const v4f*)(wp + Nn + 4);
        }
    };
    auto l_store = [&](int buf) {
        _Float16* la = lA + buf * G_ASZ;
        v8h pa;
#pragma unroll
        for (int j = 0; j < 4; ++j) { pa[j] = (_Float16)rA0[j]; pa[4 + j] = (_Float16)rA1[j]; }
        *(v8h*)&la[ar * G_LAS + ac] = pa;            // one ds_store_b128
        _Float16* lb = lB + buf * G_BSZ;
#pragma unroll
        for (int rep = 0; rep < 2; ++rep) {
            const int k = 2 * (bk + rep * 8);
#pragma unroll
            for (int j = 0; j < 4; ++j) {            // b32 pair stores
                *(unsigned*)&lb[(bn + j)     * G_LBS + k] = pk2(rB[rep][0][j], rB[rep][2][j]);
                *(unsigned*)&lb[(bn + 4 + j) * G_LBS + k] = pk2(rB[rep][1][j], rB[rep][3][j]);
            }
        }
    };
    auto compute = [&](int buf) {
        const _Float16* la = lA + buf * G_ASZ;
        const _Float16* lb = lB + buf * G_BSZ;
        v16h af[2], bf[4];
#pragma unroll
        for (int mt = 0; mt < 2; ++mt) {
            const int r  = waveM * 32 + mt * 16 + l16;
            const int ko = half * 8;   // lane<16: K{0..7,16..23}; else {8..15,24..31}
            af[mt] = mk16(&la[r * G_LAS + ko], &la[r * G_LAS + ko + 16]);
        }
#pragma unroll
        for (int nt = 0; nt < 4; ++nt) {
            const int c  = waveN * 64 + nt * 16 + l16;
            const int ko = half * 8;   // lane<16: K{0..7,16..23}; else {8..15,24..31}
            bf[nt] = mk16(&lb[c * G_LBS + ko], &lb[c * G_LBS + ko + 16]);
        }
#pragma unroll
        for (int mt = 0; mt < 2; ++mt)
#pragma unroll
            for (int nt = 0; nt < 4; ++nt)
                acc[mt][nt] = wmma16(af[mt], bf[nt], acc[mt][nt]);
    };

    // software pipeline: prefetch k0+32 globals while computing k0
    g_load(0);
    l_store(0);
    __syncthreads();
    int cur = 0;
    for (int k0 = 0; k0 < K; k0 += 32) {
        const bool more = (k0 + 32 < K);
        if (more) g_load(k0 + 32);      // issue early; consumed in l_store
        compute(cur);
        if (more) l_store(cur ^ 1);
        __syncthreads();
        cur ^= 1;
    }

    // ---- epilogue: bias (+res) (+gelu)
#pragma unroll
    for (int mt = 0; mt < 2; ++mt)
#pragma unroll
        for (int nt = 0; nt < 4; ++nt) {
            const int col = col0 + waveN * 64 + nt * 16 + l16;
            const float bcol = bias[col];
#pragma unroll
            for (int i = 0; i < 8; ++i) {
                const int row = row0 + waveM * 32 + mt * 16 + half * 8 + i;
                float v = acc[mt][nt][i] + bcol;
                if (res)  v += res[(size_t)row * Nn + col];
                if (gelu) v = 0.5f * v * (1.0f + erff(v * 0.70710678118654752f));
                out[(size_t)row * Nn + col] = v;
            }
        }
}

// ---------------------------------------------------------------------------
// Flash attention: O = softmax(Q K^T * scale) V, head dim 128, H heads packed
// in columns of D=1024. Grid (rowsQ/64, H, B); 128 threads = 4 waves; each
// wave owns 16 query rows; 32-key chunks double-buffered in LDS.
// ---------------------------------------------------------------------------
#define A_LKS 136               // lK row stride f16 (17*8: 16B aligned, spread)
#define A_LVS 40                // lVt row stride
#define A_LPS 40                // lP  row stride
#define A_KSZ (32 * A_LKS)
#define A_VSZ (128 * A_LVS)

__global__ __launch_bounds__(128, 1)
void attn_kernel(const float* __restrict__ Q, const float* __restrict__ Km,
                 const float* __restrict__ V, float* __restrict__ O,
                 int rowsQ, int rowsKV, float scale) {
    __shared__ _Float16 lK[2 * A_KSZ];       // [key][ch]
    __shared__ _Float16 lV[2 * A_VSZ];       // [ch][key] (transposed)
    __shared__ _Float16 lP[4 * 16 * A_LPS];  // per-wave [qrow][key]

    const int Dm   = 1024;
    const int tid  = threadIdx.x;
    const int wid  = tid >> 5;
    const int lane = tid & 31;
    const int half = lane >> 4;
    const int l16  = lane & 15;
    const int b    = blockIdx.z;
    const int head = blockIdx.y;
    const int chb  = head * 128;
    const size_t qrow0  = (size_t)b * rowsQ + blockIdx.x * 64 + wid * 16;
    const size_t kvrow0 = (size_t)b * rowsKV;
    _Float16* lPw = &lP[wid * 16 * A_LPS];

    // ---- preload Q fragments (held for the whole key loop)
    v16h qf[4];
    {
        const float* qp = Q + (qrow0 + l16) * Dm + chb;
#pragma unroll
        for (int kc = 0; kc < 4; ++kc) {
            const int ko = kc * 32 + half * 8;
            V16U u;
#pragma unroll
            for (int j = 0; j < 8; ++j) {
                u.e[j]     = (_Float16)qp[ko + j];
                u.e[8 + j] = (_Float16)qp[ko + 16 + j];
            }
            qf[kc] = u.v;
        }
    }

    float mrow[8], lrow[8];
    v8f o[8];
#pragma unroll
    for (int i = 0; i < 8; ++i) { mrow[i] = -3.0e38f; lrow[i] = 0.0f; }
#pragma unroll
    for (int c = 0; c < 8; ++c) o[c] = v8f_zero();

    // staging map: thread handles 2 adjacent keys x 16 channels
    const int key0 = (tid & 15) * 2;        // 0..30 (even)
    const int ch0  = (tid >> 4) * 16;       // 0..112

    v4f rK[2][4], rV[2][4];                 // [key in pair][4x v4f = 16 ch]

    auto g_load = [&](int jb) {
#pragma unroll
        for (int kk = 0; kk < 2; ++kk) {
            const float* kp = Km + (kvrow0 + jb + key0 + kk) * Dm + chb + ch0;
            const float* vp = V  + (kvrow0 + jb + key0 + kk) * Dm + chb + ch0;
#pragma unroll
            for (int q = 0; q < 4; ++q) {
                rK[kk][q] = *(const v4f*)(kp + 4 * q);
                rV[kk][q] = *(const v4f*)(vp + 4 * q);
            }
        }
    };
    auto l_store = [&](int buf) {
        _Float16* lKc = lK + buf * A_KSZ;
        _Float16* lVc = lV + buf * A_VSZ;
#pragma unroll
        for (int kk = 0; kk < 2; ++kk) {       // K rows: contiguous b128 stores
            v8h h0, h1;
#pragma unroll
            for (int j = 0; j < 4; ++j) {
                h0[j]     = (_Float16)rK[kk][0][j];
                h0[4 + j] = (_Float16)rK[kk][1][j];
                h1[j]     = (_Float16)rK[kk][2][j];
                h1[4 + j] = (_Float16)rK[kk][3][j];
            }
            *(v8h*)&lKc[(key0 + kk) * A_LKS + ch0]     = h0;
            *(v8h*)&lKc[(key0 + kk) * A_LKS + ch0 + 8] = h1;
        }
#pragma unroll
        for (int q = 0; q < 4; ++q)            // V transposed: b32 pair stores
#pragma unroll
            for (int e = 0; e < 4; ++e)
                *(unsigned*)&lVc[(ch0 + 4 * q + e) * A_LVS + key0] =
                    pk2(rV[0][q][e], rV[1][q][e]);
    };
    auto compute = [&](int buf) {
        const _Float16* lKc = lK + buf * A_KSZ;
        const _Float16* lVc = lV + buf * A_VSZ;
        // ---- S = Q K^T for 2 key subtiles (8 WMMAs)
        v8f s0 = v8f_zero(), s1 = v8f_zero();
#pragma unroll
        for (int kc = 0; kc < 4; ++kc) {
            const int ko = kc * 32 + half * 8;
            v16h b0 = mk16(&lKc[l16 * A_LKS + ko],        &lKc[l16 * A_LKS + ko + 16]);
            v16h b1 = mk16(&lKc[(16 + l16) * A_LKS + ko], &lKc[(16 + l16) * A_LKS + ko + 16]);
            s0 = wmma16(qf[kc], b0, s0);
            s1 = wmma16(qf[kc], b1, s1);
        }
        // ---- online softmax (rows on (VGPR i, lane-half); cols on lane&15)
#pragma unroll
        for (int i = 0; i < 8; ++i) {
            float a0 = s0[i] * scale, a1 = s1[i] * scale;
            float mx = fmaxf(a0, a1);
#pragma unroll
            for (int off = 1; off < 16; off <<= 1)
                mx = fmaxf(mx, __shfl_xor(mx, off, 32));
            const float mn   = fmaxf(mrow[i], mx);
            const float corr = __expf(mrow[i] - mn);
            const float p0 = __expf(a0 - mn), p1 = __expf(a1 - mn);
            float ps = p0 + p1;
#pragma unroll
            for (int off = 1; off < 16; off <<= 1)
                ps += __shfl_xor(ps, off, 32);
            lrow[i] = lrow[i] * corr + ps;
            mrow[i] = mn;
#pragma unroll
            for (int c = 0; c < 8; ++c) o[c][i] *= corr;
            const int ri = i + half * 8;
            lPw[ri * A_LPS + l16]      = (_Float16)p0;
            lPw[ri * A_LPS + 16 + l16] = (_Float16)p1;
        }
        // ---- O += P V (P reshaped via per-wave LDS into A layout; 8 WMMAs)
        v16h pa = mk16(&lPw[l16 * A_LPS + half * 8],
                       &lPw[l16 * A_LPS + 16 + half * 8]);
#pragma unroll
        for (int c = 0; c < 8; ++c) {
            const int vr = (c * 16 + l16) * A_LVS + half * 8;
            v16h vb = mk16(&lVc[vr], &lVc[vr + 16]);
            o[c] = wmma16(pa, vb, o[c]);
        }
    };

    // software pipeline over 32-key chunks
    g_load(0);
    l_store(0);
    __syncthreads();
    int cur = 0;
    for (int jb = 0; jb < rowsKV; jb += 32) {
        const bool more = (jb + 32 < rowsKV);
        if (more) g_load(jb + 32);
        compute(cur);
        if (more) l_store(cur ^ 1);
        __syncthreads();
        cur ^= 1;
    }

    // ---- normalize and write out
#pragma unroll
    for (int c = 0; c < 8; ++c)
#pragma unroll
        for (int i = 0; i < 8; ++i) {
            const size_t row = qrow0 + half * 8 + i;
            const int col = chb + c * 16 + l16;
            O[row * Dm + col] = o[c][i] / lrow[i];
        }
}

// ---------------------------------------------------------------------------
// Host orchestration
// ---------------------------------------------------------------------------
extern "C" void kernel_launch(void* const* d_in, const int* in_sizes, int n_in,
                              void* d_out, int out_size, void* d_ws, size_t ws_size,
                              hipStream_t stream) {
    (void)in_sizes; (void)n_in; (void)out_size; (void)ws_size;
    const int B = 8, M = 4096, C = 512, N = 512, D = 1024, DW = 4096, H = 8, L = 6;
    const int BN = B * N;       // 4096
    const int BM = B * M;       // 32768
    const float scale = 0.08838834764831845f;  // 128^-0.5

    const float* x        = (const float*)d_in[0];
    const float* latent   = (const float*)d_in[1];
    const float* ca_qn_w  = (const float*)d_in[2];
    const float* ca_qn_b  = (const float*)d_in[3];
    const float* ca_kvn_w = (const float*)d_in[4];
    const float* ca_kvn_b = (const float*)d_in[5];
    const float* ca_wq = (const float*)d_in[6];
    const float* ca_bq = (const float*)d_in[7];
    const float* ca_wk = (const float*)d_in[8];
    const float* ca_bk = (const float*)d_in[9];
    const float* ca_wv = (const float*)d_in[10];
    const float* ca_bv = (const float*)d_in[11];
    const float* ca_wo = (const float*)d_in[12];
    const float* ca_bo = (const float*)d_in[13];
    const float* ca_ln_w = (const float*)d_in[14];
    const float* ca_ln_b = (const float*)d_in[15];
    const float* ca_w1 = (const float*)d_in[16];
    const float* ca_b1 = (const float*)d_in[17];
    const float* ca_w2 = (const float*)d_in[18];
    const float* ca_b2 = (const float*)d_in[19];
    const float* sa_n_w  = (const float*)d_in[20];
    const float* sa_n_b  = (const float*)d_in[21];
    const float* sa_wq = (const float*)d_in[22];
    const float* sa_bq = (const float*)d_in[23];
    const float* sa_wk = (const float*)d_in[24];
    const float* sa_bk = (const float*)d_in[25];
    const float* sa_wv = (const float*)d_in[26];
    const float* sa_bv = (const float*)d_in[27];
    const float* sa_wo = (const float*)d_in[28];
    const float* sa_bo = (const float*)d_in[29];
    const float* sa_ln_w = (const float*)d_in[30];
    const float* sa_ln_b = (const float*)d_in[31];
    const float* sa_w1 = (const float*)d_in[32];
    const float* sa_b1 = (const float*)d_in[33];
    const float* sa_w2 = (const float*)d_in[34];
    const float* sa_b2 = (const float*)d_in[35];

    // workspace carve (fp32)
    char* wsb = (char*)d_ws;
    size_t off = 0;
    auto nb = [&](size_t nelem) {
        void* p = wsb + off;
        off += (nelem * 4 + 255) & ~(size_t)255;
        return (float*)p;
    };
    float* hA  = nb((size_t)BN * D);
    float* hB  = nb((size_t)BN * D);
    float* lnO = nb((size_t)BN * D);
    float* qB  = nb((size_t)BN * D);
    float* kB  = nb((size_t)BM * D);
    float* vB  = nb((size_t)BM * D);
    float* oB  = nb((size_t)BN * D);
    float* hid = nb((size_t)BN * DW);
    float* xln = nb((size_t)BM * C);

    const dim3 gD(D / 256, BN / 64);     // [BN,*]@[*,D]
    const dim3 gDW(DW / 256, BN / 64);   // [BN,*]@[*,DW]
    const dim3 gKV(D / 256, BM / 64);    // [BM,C]@[C,D]
    const dim3 gAtt(N / 64, H, B);

    // ---- cross-attention block ----
    {
        const int tot = BN * D;
        bcast_kernel<<<(tot + 255) / 256, 256, 0, stream>>>(latent, hA, N * D, tot);
    }
    ln_kernel<<<BN, 256, 0, stream>>>(hA, lnO, ca_qn_w, ca_qn_b, D);
    ln_kernel<<<BM, 256, 0, stream>>>(x, xln, ca_kvn_w, ca_kvn_b, C);
    gemm_kernel<<<gD, 256, 0, stream>>>(lnO, ca_wq, ca_bq, nullptr, qB, D, D, 0);
    gemm_kernel<<<gKV, 256, 0, stream>>>(xln, ca_wk, ca_bk, nullptr, kB, C, D, 0);
    gemm_kernel<<<gKV, 256, 0, stream>>>(xln, ca_wv, ca_bv, nullptr, vB, C, D, 0);
    attn_kernel<<<gAtt, 128, 0, stream>>>(qB, kB, vB, oB, N, M, scale);
    gemm_kernel<<<gD, 256, 0, stream>>>(oB, ca_wo, ca_bo, hA, hB, D, D, 0);  // +res
    ln_kernel<<<BN, 256, 0, stream>>>(hB, lnO, ca_ln_w, ca_ln_b, D);
    gemm_kernel<<<gDW, 256, 0, stream>>>(lnO, ca_w1, ca_b1, nullptr, hid, D, DW, 1); // GELU
    gemm_kernel<<<gD, 256, 0, stream>>>(hid, ca_w2, ca_b2, hB, hA, DW, D, 0);        // +res

    // ---- self-attention layers (residual stream stays in hA; hB is "a") ----
    for (int i = 0; i < L; ++i) {
        const size_t dd  = (size_t)i * D * D;
        const size_t dv  = (size_t)i * D;
        const size_t dw1 = (size_t)i * D * DW;
        const size_t dbw = (size_t)i * DW;
        ln_kernel<<<BN, 256, 0, stream>>>(hA, lnO, sa_n_w + dv, sa_n_b + dv, D);
        gemm_kernel<<<gD, 256, 0, stream>>>(lnO, sa_wq + dd, sa_bq + dv, nullptr, qB, D, D, 0);
        gemm_kernel<<<gD, 256, 0, stream>>>(lnO, sa_wk + dd, sa_bk + dv, nullptr, kB, D, D, 0);
        gemm_kernel<<<gD, 256, 0, stream>>>(lnO, sa_wv + dd, sa_bv + dv, nullptr, vB, D, D, 0);
        attn_kernel<<<gAtt, 128, 0, stream>>>(qB, kB, vB, oB, N, N, scale);
        gemm_kernel<<<gD, 256, 0, stream>>>(oB, sa_wo + dd, sa_bo + dv, hA, hB, D, D, 0); // a
        ln_kernel<<<BN, 256, 0, stream>>>(hB, lnO, sa_ln_w + dv, sa_ln_b + dv, D);
        gemm_kernel<<<gDW, 256, 0, stream>>>(lnO, sa_w1 + dw1, sa_b1 + dbw, nullptr, hid, D, DW, 1);
        float* dst = (i == L - 1) ? (float*)d_out : hA;  // h = mlp + a
        gemm_kernel<<<gD, 256, 0, stream>>>(hid, sa_w2 + dw1, sa_b2 + dv, hB, dst, DW, D, 0);
    }
}
